// DotProductAttention_44873818308757
// MI455X (gfx1250) — matrix operand order repaired and checked by reading, not compiled, not run on hardware
//
#include <hip/hip_runtime.h>

// Problem constants (match the reference harness).
#define BB 16
#define LL 2048
#define DD 128

typedef __attribute__((ext_vector_type(16))) __bf16        v16bf;
typedef __attribute__((ext_vector_type(8)))  float         v8f;
typedef __attribute__((ext_vector_type(8)))  unsigned int  v8u;

static constexpr float LOG2E  = 1.4426950408889634f;
static constexpr float QSCALE = 0.08838834764831845f * LOG2E; // (1/sqrt(128)) * log2(e)

// Flash-attention: 1 wave = 16 queries, 4 waves/block, loop over 32-key tiles.
// WMMA layouts (wave32); C/D per CDNA5 ISA, A/B in this code's k order:
//   A (16x32 bf16): lane holds row m=lane&15; VGPR j holds K = (lane>>4)*16 + 2*j .. +1 (contiguous order)
//   B (32x16 bf16): lane holds col n=lane&15; VGPR j holds K = (lane>>4)*16 + 2*j .. +1 (contiguous order)
//   C/D (16x16 f32): lane holds col n=lane&15; VGPR r holds row m = r + 8*(lane>>4)
__global__ __launch_bounds__(128) void attn_flash_bf16_wmma(
    const float* __restrict__ Q, const float* __restrict__ K,
    const float* __restrict__ V, const int* __restrict__ vlens,
    float* __restrict__ Out)
{
    // Per-wave P staging tile: 16 rows x 34 bf16 (pad 32->34 to avoid bank conflicts)
    __shared__ __align__(16) __bf16 Plds[4][16 * 34];

    const int lane = threadIdx.x & 31;
    const int wv   = threadIdx.x >> 5;
    const int n16  = lane & 15;   // N / column index within WMMA tile
    const int h    = lane >> 4;   // half-wave index
    const int b    = blockIdx.y;
    const int qbase = (blockIdx.x * 4 + wv) * 16;

    const float* Qp = Q + ((size_t)b * LL + qbase) * DD;
    const float* Kp = K + (size_t)b * LL * DD;
    const float* Vp = V + (size_t)b * LL * DD;
    float*       Op = Out + ((size_t)b * LL + qbase) * DD;
    const int vlen  = vlens[b];

    if (vlen == 0) {
        // All keys masked -> softmax over uniform -1e6 scores -> uniform weights:
        // every output row is the mean of V over all 2048 keys.
        const int d0 = lane * 4; // 32 lanes x 4 floats = 128 dims
        float4 acc = {0.f, 0.f, 0.f, 0.f};
        for (int k = 0; k < LL; ++k) {
            const float4 v = *(const float4*)(Vp + (size_t)k * DD + d0);
            acc.x += v.x; acc.y += v.y; acc.z += v.z; acc.w += v.w;
        }
        const float inv = 1.0f / (float)LL;
        acc.x *= inv; acc.y *= inv; acc.z *= inv; acc.w *= inv;
        for (int m = 0; m < 16; ++m)
            *(float4*)(Op + (size_t)m * DD + d0) = acc;
        return;
    }

    // ---- Load Q tile once into A-layout registers (scale folded in) ----
    v16bf qa[4];
#pragma unroll
    for (int ds = 0; ds < 4; ++ds) {
#pragma unroll
        for (int j = 0; j < 8; ++j) {
            const int kk = ds * 32 + h * 16 + 2 * j;
            const float2 f = *(const float2*)(Qp + (size_t)n16 * DD + kk);
            qa[ds][2 * j]     = (__bf16)(f.x * QSCALE);
            qa[ds][2 * j + 1] = (__bf16)(f.y * QSCALE);
        }
    }

    // ---- Accumulators ----
    v8f o[8];
#pragma unroll
    for (int nd = 0; nd < 8; ++nd) o[nd] = (v8f){0, 0, 0, 0, 0, 0, 0, 0};
    v8f lacc = (v8f){0, 0, 0, 0, 0, 0, 0, 0}; // row sums via ones-column WMMA

    v16bf onesb;
#pragma unroll
    for (int e = 0; e < 16; ++e) onesb[e] = (__bf16)1.0f;

    // Early exit: key tiles fully beyond vlen contribute exp(-1e6-..)==0 exactly.
    const int nkt = (vlen + 31) >> 5;
    for (int kt = 0; kt < nkt; ++kt) {
        const int kt32 = kt * 32;

        // Prefetch a slice of next K tile (one instruction, spread across lanes).
        if (kt + 1 < nkt)
            __builtin_prefetch(Kp + (size_t)(kt32 + 32) * DD + lane * 32, 0, 1);

        // ---- S' = (Q*scale*log2e) . K^T : two 16x16 C-tiles, K-dim = 128 ----
        v8f s0 = (v8f){0, 0, 0, 0, 0, 0, 0, 0};
        v8f s1 = (v8f){0, 0, 0, 0, 0, 0, 0, 0};
#pragma unroll
        for (int ds = 0; ds < 4; ++ds) {
            v16bf kb;
#pragma unroll
            for (int j = 0; j < 8; ++j) {
                const int kk = ds * 32 + h * 16 + 2 * j;
                const float2 f =
                    *(const float2*)(Kp + (size_t)(kt32 + n16) * DD + kk);
                kb[2 * j] = (__bf16)f.x; kb[2 * j + 1] = (__bf16)f.y;
            }
            s0 = __builtin_amdgcn_wmma_f32_16x16x32_bf16(
                false, qa[ds], false, kb, (short)0, s0, false, false);
#pragma unroll
            for (int j = 0; j < 8; ++j) {
                const int kk = ds * 32 + h * 16 + 2 * j;
                const float2 f =
                    *(const float2*)(Kp + (size_t)(kt32 + 16 + n16) * DD + kk);
                kb[2 * j] = (__bf16)f.x; kb[2 * j + 1] = (__bf16)f.y;
            }
            s1 = __builtin_amdgcn_wmma_f32_16x16x32_bf16(
                false, qa[ds], false, kb, (short)0, s1, false, false);
        }

        // ---- mask + exp2, scatter P (bf16) to LDS in row-major [m][k] ----
        const bool v0 = (kt32 + n16) < vlen;
        const bool v1 = (kt32 + 16 + n16) < vlen;
#pragma unroll
        for (int r = 0; r < 8; ++r) {
            const int m = r + 8 * h;
            const float p0 = v0 ? exp2f(s0[r]) : 0.0f;
            const float p1 = v1 ? exp2f(s1[r]) : 0.0f;
            Plds[wv][m * 34 + n16]      = (__bf16)p0;
            Plds[wv][m * 34 + 16 + n16] = (__bf16)p1;
        }
        // Same-wave cross-lane LDS round trip: order stores before loads.
        __asm__ volatile("s_wait_dscnt 0" ::: "memory");

        // ---- Re-read P in A-layout (packed bf16 pairs are dword loads) ----
        v8u pau;
#pragma unroll
        for (int j = 0; j < 8; ++j) {
            const int kk = h * 16 + 2 * j;
            pau[j] = *(const unsigned int*)&Plds[wv][n16 * 34 + kk];
        }
        const v16bf pa = __builtin_bit_cast(v16bf, pau);

        // Row sums: one ones-column WMMA instead of shuffle reductions.
        lacc = __builtin_amdgcn_wmma_f32_16x16x32_bf16(
            false, pa, false, onesb, (short)0, lacc, false, false);

        // ---- O += P . V  (8 output column tiles of 16) ----
#pragma unroll
        for (int nd = 0; nd < 8; ++nd) {
            v16bf vb;
#pragma unroll
            for (int j = 0; j < 8; ++j) {
                const int key = kt32 + h * 16 + 2 * j;
                const int d   = nd * 16 + n16;
                vb[2 * j]     = (__bf16)Vp[(size_t)key * DD + d];
                vb[2 * j + 1] = (__bf16)Vp[(size_t)(key + 1) * DD + d];
            }
            o[nd] = __builtin_amdgcn_wmma_f32_16x16x32_bf16(
                false, pa, false, vb, (short)0, o[nd], false, false);
        }
    }

    // ---- Normalize and store (C-layout: lane n = column, VGPR r -> row) ----
#pragma unroll
    for (int r = 0; r < 8; ++r) {
        const float inv = 1.0f / lacc[r]; // lacc[r] = row sum (vlen>0 => >0)
        const int m = r + 8 * h;
#pragma unroll
        for (int nd = 0; nd < 8; ++nd)
            Op[(size_t)m * DD + nd * 16 + n16] = o[nd][r] * inv;
    }
}

extern "C" void kernel_launch(void* const* d_in, const int* in_sizes, int n_in,
                              void* d_out, int out_size, void* d_ws, size_t ws_size,
                              hipStream_t stream) {
    (void)in_sizes; (void)n_in; (void)d_ws; (void)ws_size; (void)out_size;
    const float* Q  = (const float*)d_in[0];
    const float* K  = (const float*)d_in[1];
    const float* V  = (const float*)d_in[2];
    const int*   vl = (const int*)d_in[3];
    float*       O  = (float*)d_out;

    dim3 grid(LL / 64, BB, 1); // 64 queries per block (4 waves x 16)
    attn_flash_bf16_wmma<<<grid, 128, 0, stream>>>(Q, K, V, vl, O);
}
